// NSMCell_17789754540886
// MI455X (gfx1250) — matrix operand order repaired and checked by reading, not compiled, not run on hardware
//
#include <hip/hip_runtime.h>
#include <hip/hip_bf16.h>

// ---------------------------------------------------------------------------
// NSMCell for MI455X (gfx1250): bf16 WMMA GEMMs with fused epilogues.
//   D = 256 assumed; wave32; 256-thread blocks = 8 waves.
//   64-row tiles: 8 waves x 2 col-tiles x 4 M-subtiles per block.
// ---------------------------------------------------------------------------

typedef __attribute__((ext_vector_type(16))) __bf16 v16bf;
typedef __attribute__((ext_vector_type(8)))  float  v8f;

#define DD 256
#define TM 64      // rows per block tile
#define KC 32      // K per WMMA

union FragU { uint4 q[2]; v16bf v; };

__device__ __forceinline__ unsigned short f2bf(float f) {
  unsigned u = __float_as_uint(f);
  unsigned r = u + 0x7FFFu + ((u >> 16) & 1u);   // round-to-nearest-even
  return (unsigned short)(r >> 16);
}

__device__ __forceinline__ float elu1(float x) {
  return x > 0.0f ? x : (expf(x) - 1.0f);
}

// ---------------------------------------------------------------------------
__global__ void zero_f32(float* __restrict__ p, int n) {
  int i = blockIdx.x * blockDim.x + threadIdx.x;
  if (i < n) p[i] = 0.0f;
}

// ---------------------------------------------------------------------------
// W_eff[b][d][k] = sum_p sim[b,p] * W_p[d][k]  (bf16), plus W_edge -> bf16.
// One thread per (d,k); loops over all B graphs (W_p read once, stores coalesced).
// ---------------------------------------------------------------------------
__global__ __launch_bounds__(256)
void prep_weights(const float* __restrict__ Wp, const float* __restrict__ sim,
                  const float* __restrict__ We,
                  unsigned short* __restrict__ Weff,
                  unsigned short* __restrict__ Wedge,
                  int B, int P) {
  int dk = blockIdx.x * blockDim.x + threadIdx.x;
  if (dk >= DD * DD) return;
  float wreg[8];
  #pragma unroll
  for (int p = 0; p < 8; ++p) wreg[p] = 0.0f;
  for (int p = 0; p < P && p < 8; ++p) wreg[p] = Wp[(size_t)p * DD * DD + dk];
  for (int b = 0; b < B; ++b) {
    float acc = 0.0f;
    for (int p = 0; p < P && p < 8; ++p) acc += sim[b * P + p] * wreg[p];
    Weff[(size_t)b * DD * DD + dk] = f2bf(acc);
  }
  Wedge[dk] = f2bf(We[dk]);
}

// ---------------------------------------------------------------------------
// Shared helper: stage a TM x 256 f32 strip into LDS as bf16 (zero-padded).
// ---------------------------------------------------------------------------
__device__ __forceinline__ void stage_strip(const float* __restrict__ src,
                                            unsigned short (*At)[DD],
                                            int row0, int nrows_total, int tid) {
  int rb = tid >> 4;            // 0..15
  int k0 = (tid & 15) << 4;     // 0..240
  #pragma unroll
  for (int i = 0; i < TM / 16; ++i) {
    int row = rb + i * 16;
    int n   = row0 + row;
    unsigned short tmp[16];
    if (n < nrows_total) {
      const float4* s = (const float4*)(src + (size_t)n * DD + k0);
      #pragma unroll
      for (int q = 0; q < 4; ++q) {
        float4 f = s[q];
        tmp[q*4+0] = f2bf(f.x); tmp[q*4+1] = f2bf(f.y);
        tmp[q*4+2] = f2bf(f.z); tmp[q*4+3] = f2bf(f.w);
      }
    } else {
      #pragma unroll
      for (int q = 0; q < 16; ++q) tmp[q] = 0;
    }
    *(uint4*)&At[row][k0]     = *(uint4*)&tmp[0];
    *(uint4*)&At[row][k0 + 8] = *(uint4*)&tmp[8];
  }
}

// ---------------------------------------------------------------------------
// Node path: state_logits[n] = sum_d elu(instr[b,d]*(W_eff[b]@attr_n)[d])*w_ns[d]
// node_indices sorted -> graphs form contiguous runs inside a tile; loop over
// runs, mask rows per 16-row M-subtile, skip empty subtiles (uniform branches).
// ---------------------------------------------------------------------------
__global__ __launch_bounds__(256)
void node_state_kernel(const float* __restrict__ attrs,
                       const unsigned short* __restrict__ Weff,
                       const float* __restrict__ instr,
                       const float* __restrict__ w_ns,
                       const int* __restrict__ nidx,
                       float* __restrict__ state_logits, int N) {
  __shared__ __align__(16) unsigned short At[TM][DD];
  __shared__ int   bi[TM];
  __shared__ float accR[TM];

  const int tid  = threadIdx.x;
  const int lane = tid & 31;
  const int wid  = tid >> 5;
  const int half = lane >> 4;
  const int l15  = lane & 15;
  const int n0   = blockIdx.x * TM;

  if (tid < TM) {
    int n = n0 + tid;
    bi[tid]   = (n < N) ? nidx[n] : -1;
    accR[tid] = 0.0f;
  }
  stage_strip(attrs, At, n0, N, tid);
  __syncthreads();

  v8f c[TM / 16][2];
  #pragma unroll
  for (int m = 0; m < TM / 16; ++m) { c[m][0] = (v8f){}; c[m][1] = (v8f){}; }

  const int col0 = (wid * 2 + 0) * 16;
  const int col1 = (wid * 2 + 1) * 16;

  // run loop: contiguous rows sharing a graph id
  int r0 = 0;
  while (r0 < TM) {
    int pb = bi[r0];
    int r1 = r0 + 1;
    while (r1 < TM && bi[r1] == pb) ++r1;
    if (pb < 0) { r0 = r1; continue; }     // tail rows
    int len = r1 - r0;
    unsigned long long pm =
        (len >= 64 ? ~0ull : ((1ull << len) - 1ull)) << r0;
    const unsigned short* Wb = Weff + (size_t)pb * DD * DD;

    for (int kc = 0; kc < DD; kc += KC) {
      FragU b0, b1;   // B 32x16: lane<16 K 0..15, lane>=16 K 16..31 (contiguous)
      {
        const uint4* p = (const uint4*)(Wb + (size_t)(col0 + l15) * DD + kc + half * 16);
        b0.q[0] = p[0]; b0.q[1] = p[1];
      }
      {
        const uint4* p = (const uint4*)(Wb + (size_t)(col1 + l15) * DD + kc + half * 16);
        b1.q[0] = p[0]; b1.q[1] = p[1];
      }
      #pragma unroll
      for (int m = 0; m < TM / 16; ++m) {
        if (((pm >> (m * 16)) & 0xFFFFull) == 0ull) continue;  // uniform skip
        FragU a;        // A 16x32 bf16: lane<16 K 0..15, lane>=16 K 16..31 (contiguous)
        if ((pm >> (m * 16 + l15)) & 1ull) {
          a.q[0] = *(const uint4*)&At[m * 16 + l15][kc + half * 16];
          a.q[1] = *(const uint4*)&At[m * 16 + l15][kc + half * 16 + 8];
        } else {
          a.q[0] = make_uint4(0, 0, 0, 0);
          a.q[1] = make_uint4(0, 0, 0, 0);
        }
        c[m][0] = __builtin_amdgcn_wmma_f32_16x16x32_bf16(false, a.v, false, b0.v,
                                                          (short)0, c[m][0], false, false);
        c[m][1] = __builtin_amdgcn_wmma_f32_16x16x32_bf16(false, a.v, false, b1.v,
                                                          (short)0, c[m][1], false, false);
      }
    }
    r0 = r1;
  }

  // epilogue: elu(instr * C) * w_ns, reduce across d per row
  #pragma unroll
  for (int t = 0; t < 2; ++t) {
    int d = (t ? col1 : col0) + l15;
    float wv = w_ns[d];
    #pragma unroll
    for (int m = 0; m < TM / 16; ++m) {
      #pragma unroll
      for (int j = 0; j < 8; ++j) {
        int row = m * 16 + j + half * 8;  // C: VGPR j -> M=j (lo half) / j+8 (hi)
        int b = bi[row];
        float iv = (b >= 0) ? instr[b * DD + d] : 0.0f;
        float v  = elu1(iv * c[m][t][j]) * wv;
        v += __shfl_xor(v, 1);
        v += __shfl_xor(v, 2);
        v += __shfl_xor(v, 4);
        v += __shfl_xor(v, 8);
        if (l15 == 0) atomicAdd(&accR[row], v);
      }
    }
  }
  __syncthreads();
  if (tid < TM) {
    int n = n0 + tid;
    if (n < N) state_logits[n] = accR[tid];
  }
}

// ---------------------------------------------------------------------------
// Edge path: t_e = sum_d elu(instr[eb,d]*(edge_attr@W_edge^T)[d])*w_rel[d];
// rel_logits[dst] += dist[src] * t_e  (scatter collapsed to one scalar/edge).
// ---------------------------------------------------------------------------
__global__ __launch_bounds__(256)
void edge_rel_kernel(const float* __restrict__ eattrs,
                     const unsigned short* __restrict__ Wedge,
                     const float* __restrict__ instr,
                     const float* __restrict__ w_rel,
                     const int* __restrict__ ebi,
                     const int* __restrict__ esrc,
                     const int* __restrict__ edst,
                     const float* __restrict__ dist,
                     float* __restrict__ rel_logits, int E) {
  __shared__ __align__(16) unsigned short At[TM][DD];
  __shared__ int   eb[TM];
  __shared__ int   dn[TM];
  __shared__ float df[TM];
  __shared__ float accR[TM];

  const int tid  = threadIdx.x;
  const int lane = tid & 31;
  const int wid  = tid >> 5;
  const int half = lane >> 4;
  const int l15  = lane & 15;
  const int e0   = blockIdx.x * TM;

  if (tid < TM) {
    int e = e0 + tid;
    if (e < E) { eb[tid] = ebi[e]; dn[tid] = edst[e]; df[tid] = dist[esrc[e]]; }
    else       { eb[tid] = -1;     dn[tid] = -1;      df[tid] = 0.0f; }
    accR[tid] = 0.0f;
  }
  stage_strip(eattrs, At, e0, E, tid);
  __syncthreads();

  v8f c[TM / 16][2];
  #pragma unroll
  for (int m = 0; m < TM / 16; ++m) { c[m][0] = (v8f){}; c[m][1] = (v8f){}; }

  const int col0 = (wid * 2 + 0) * 16;
  const int col1 = (wid * 2 + 1) * 16;

  for (int kc = 0; kc < DD; kc += KC) {
    FragU b0, b1;
    {
      const uint4* p = (const uint4*)(Wedge + (size_t)(col0 + l15) * DD + kc + half * 16);
      b0.q[0] = p[0]; b0.q[1] = p[1];
    }
    {
      const uint4* p = (const uint4*)(Wedge + (size_t)(col1 + l15) * DD + kc + half * 16);
      b1.q[0] = p[0]; b1.q[1] = p[1];
    }
    #pragma unroll
    for (int m = 0; m < TM / 16; ++m) {
      FragU a;
      a.q[0] = *(const uint4*)&At[m * 16 + l15][kc + half * 16];
      a.q[1] = *(const uint4*)&At[m * 16 + l15][kc + half * 16 + 8];
      c[m][0] = __builtin_amdgcn_wmma_f32_16x16x32_bf16(false, a.v, false, b0.v,
                                                        (short)0, c[m][0], false, false);
      c[m][1] = __builtin_amdgcn_wmma_f32_16x16x32_bf16(false, a.v, false, b1.v,
                                                        (short)0, c[m][1], false, false);
    }
  }

  #pragma unroll
  for (int t = 0; t < 2; ++t) {
    int d = (t ? col1 : col0) + l15;
    float wv = w_rel[d];
    #pragma unroll
    for (int m = 0; m < TM / 16; ++m) {
      #pragma unroll
      for (int j = 0; j < 8; ++j) {
        int row = m * 16 + j + half * 8;
        int b = eb[row];
        float iv = (b >= 0) ? instr[b * DD + d] : 0.0f;
        float v  = elu1(iv * c[m][t][j]) * wv;
        v += __shfl_xor(v, 1);
        v += __shfl_xor(v, 2);
        v += __shfl_xor(v, 4);
        v += __shfl_xor(v, 8);
        if (l15 == 0) atomicAdd(&accR[row], v);
      }
    }
  }
  __syncthreads();
  if (tid < TM && dn[tid] >= 0) {
    atomicAdd(&rel_logits[dn[tid]], df[tid] * accR[tid]);
  }
}

// ---------------------------------------------------------------------------
// Per-graph segment softmax over both logit arrays + final blend.
// ---------------------------------------------------------------------------
__device__ __forceinline__ int lbound(const int* __restrict__ a, int n, int key) {
  int lo = 0, hi = n;
  while (lo < hi) { int mid = (lo + hi) >> 1; if (a[mid] < key) lo = mid + 1; else hi = mid; }
  return lo;
}

__device__ float block_reduce(float v, bool do_max, float* scratch) {
  #pragma unroll
  for (int off = 16; off; off >>= 1) {
    float o = __shfl_xor(v, off);
    v = do_max ? fmaxf(v, o) : (v + o);
  }
  int wid = threadIdx.x >> 5, lane = threadIdx.x & 31;
  if (lane == 0) scratch[wid] = v;
  __syncthreads();
  if (threadIdx.x == 0) {
    float r = scratch[0];
    for (int w = 1; w < 8; ++w) r = do_max ? fmaxf(r, scratch[w]) : (r + scratch[w]);
    scratch[8] = r;
  }
  __syncthreads();
  float r = scratch[8];
  __syncthreads();
  return r;
}

__global__ __launch_bounds__(256)
void softmax_combine(const float* __restrict__ state_logits,
                     const float* __restrict__ rel_logits,
                     const int* __restrict__ nidx,
                     const float* __restrict__ relsim,
                     float* __restrict__ out, int N) {
  __shared__ int rng[2];
  __shared__ float scratch[9];
  int b = blockIdx.x;
  if (threadIdx.x == 0) {
    rng[0] = lbound(nidx, N, b);
    rng[1] = lbound(nidx, N, b + 1);
  }
  __syncthreads();
  int lo = rng[0], hi = rng[1];

  float ms = -3.4e38f, mr = -3.4e38f;
  for (int i = lo + threadIdx.x; i < hi; i += blockDim.x) {
    ms = fmaxf(ms, state_logits[i]);
    mr = fmaxf(mr, rel_logits[i]);
  }
  ms = block_reduce(ms, true, scratch);
  mr = block_reduce(mr, true, scratch);

  float ss = 0.0f, sr = 0.0f;
  for (int i = lo + threadIdx.x; i < hi; i += blockDim.x) {
    ss += expf(state_logits[i] - ms);
    sr += expf(rel_logits[i] - mr);
  }
  ss = block_reduce(ss, false, scratch);
  sr = block_reduce(sr, false, scratch);

  float r = relsim[b];
  float inv_ss = (ss > 0.0f) ? 1.0f / ss : 0.0f;
  float inv_sr = (sr > 0.0f) ? 1.0f / sr : 0.0f;
  for (int i = lo + threadIdx.x; i < hi; i += blockDim.x) {
    float ns = expf(state_logits[i] - ms) * inv_ss;
    float nr = expf(rel_logits[i] - mr) * inv_sr;
    out[i] = r * nr + (1.0f - r) * ns;
  }
}

// ---------------------------------------------------------------------------
extern "C" void kernel_launch(void* const* d_in, const int* in_sizes, int n_in,
                              void* d_out, int out_size, void* d_ws, size_t ws_size,
                              hipStream_t stream) {
  const float* instr  = (const float*)d_in[0];
  const float* dist   = (const float*)d_in[1];
  const float* sim    = (const float*)d_in[2];
  const float* relsim = (const float*)d_in[3];
  const float* nattr  = (const float*)d_in[4];
  const float* eattr  = (const float*)d_in[5];
  const float* Wp     = (const float*)d_in[6];
  const float* We     = (const float*)d_in[7];
  const float* w_ns   = (const float*)d_in[8];
  const float* w_rs   = (const float*)d_in[9];
  const int*   nidx   = (const int*)d_in[10];
  const int*   ebi    = (const int*)d_in[11];
  const int*   eidx   = (const int*)d_in[12];

  const int B = in_sizes[3];
  const int N = in_sizes[1];
  const int P = in_sizes[2] / (B > 0 ? B : 1);
  const int E = in_sizes[11];
  const int* esrc = eidx;
  const int* edst = eidx + E;

  // workspace layout
  char* ws = (char*)d_ws;
  size_t off = 0;
  unsigned short* Weff = (unsigned short*)(ws + off); off += (size_t)B * DD * DD * 2;
  unsigned short* Wedge = (unsigned short*)(ws + off); off += (size_t)DD * DD * 2;
  off = (off + 255) & ~(size_t)255;
  float* state_logits = (float*)(ws + off); off += (size_t)N * 4;
  off = (off + 255) & ~(size_t)255;
  float* rel_logits = (float*)(ws + off);

  zero_f32<<<(N + 255) / 256, 256, 0, stream>>>(rel_logits, N);
  prep_weights<<<(DD * DD + 255) / 256, 256, 0, stream>>>(Wp, sim, We, Weff, Wedge, B, P);
  node_state_kernel<<<(N + TM - 1) / TM, 256, 0, stream>>>(
      nattr, Weff, instr, w_ns, nidx, state_logits, N);
  edge_rel_kernel<<<(E + TM - 1) / TM, 256, 0, stream>>>(
      eattr, Wedge, instr, w_rs, ebi, esrc, edst, dist, rel_logits, E);
  softmax_combine<<<B, 256, 0, stream>>>(state_logits, rel_logits, nidx, relsim,
                                         (float*)d_out, N);
}
